// MultiHeadAttention_9672266350873
// MI455X (gfx1250) — matrix operand order repaired and checked by reading, not compiled, not run on hardware
//
#include <hip/hip_runtime.h>
#include <hip/hip_bf16.h>

typedef __attribute__((ext_vector_type(16))) __bf16 bf16x16;
typedef __attribute__((ext_vector_type(8)))  float  f32x8;
typedef int v4i __attribute__((vector_size(16)));   // matches async builtin param type

#define AS1 __attribute__((address_space(1)))
#define AS3 __attribute__((address_space(3)))

#if defined(__has_builtin)
#if __has_builtin(__builtin_amdgcn_global_load_async_to_lds_b128) && \
    __has_builtin(__builtin_amdgcn_s_wait_asynccnt)
#define USE_ASYNC_LDS 1
#endif
#endif
#ifndef USE_ASYNC_LDS
#define USE_ASYNC_LDS 0
#endif

// 16-byte global -> LDS copy: async (no VGPR staging, ASYNCcnt) when available.
__device__ __forceinline__ void copy16_g2l(__bf16* ldst, const __bf16* gsrc) {
#if USE_ASYNC_LDS
  __builtin_amdgcn_global_load_async_to_lds_b128(
      (AS1 v4i*)(unsigned long long)gsrc,
      (AS3 v4i*)(unsigned long long)ldst, 0, 0);
#else
  *(uint4*)ldst = *(const uint4*)gsrc;
#endif
}
__device__ __forceinline__ void wait_async_copies() {
#if USE_ASYNC_LDS
  __builtin_amdgcn_s_wait_asynccnt(0);
#endif
}

union Frag {
  unsigned u[8];
  bf16x16  v;
};

// A-fragment (16-bit, 16x32 MxK), source row-major [m][k]:
// lane l -> M = l&15, k-group = l>>4; u[0..7]=k{0..15}+16kg
__device__ __forceinline__ void lds_load_afrag(const unsigned* base, int row0,
                                               int stride_u32, int kbase_u32, Frag& f) {
  const int lane = threadIdx.x & 31;
  const unsigned* p = base + (row0 + (lane & 15)) * stride_u32 + kbase_u32 + ((lane >> 4) << 3);
  f.u[0] = p[0]; f.u[1] = p[1]; f.u[2] = p[2];  f.u[3] = p[3];
  f.u[4] = p[4]; f.u[5] = p[5]; f.u[6] = p[6]; f.u[7] = p[7];
}

// B-fragment (16-bit, 32x16 KxN) from N-major [n][k] storage (contiguous pairs):
__device__ __forceinline__ void lds_load_bfrag(const unsigned* base, int n0,
                                               int stride_u32, int kbase_u32, Frag& f) {
  const int lane = threadIdx.x & 31;
  const unsigned* p = base + (n0 + (lane & 15)) * stride_u32 + kbase_u32 + ((lane >> 4) << 3);
#pragma unroll
  for (int j = 0; j < 8; ++j) f.u[j] = p[j];
}

__device__ __forceinline__ f32x8 wmma_bf16(const Frag& a, const Frag& b, f32x8 c) {
  return __builtin_amdgcn_wmma_f32_16x16x32_bf16(false, a.v, false, b.v,
                                                 (short)0, c, false, false);
}

// ---------------------------------------------------------------------------
// fp32 -> bf16 flat pack (4 elements / thread)
// ---------------------------------------------------------------------------
__global__ __launch_bounds__(256) void f32_to_bf16x4(const float* __restrict__ in,
                                                     __bf16* __restrict__ out, int n4) {
  int i = blockIdx.x * 256 + threadIdx.x;
  if (i < n4) {
    float4 f = ((const float4*)in)[i];
    union { __bf16 h[4]; uint2 u; } pk;
    pk.h[0] = (__bf16)f.x; pk.h[1] = (__bf16)f.y;
    pk.h[2] = (__bf16)f.z; pk.h[3] = (__bf16)f.w;
    ((uint2*)out)[i] = pk.u;
  }
}

// ---------------------------------------------------------------------------
// fp32 [R][Cc] -> bf16 [Cc][R] transpose-convert (32x32 LDS tiles).
// One-time cost so every GEMM B-fragment becomes contiguous ds_load_b128.
// ---------------------------------------------------------------------------
__global__ __launch_bounds__(256) void transpose_f32_to_bf16(
    const float* __restrict__ in, __bf16* __restrict__ out, int R, int Cc) {
  __shared__ __bf16 sT[32 * 33];
  const int r0 = blockIdx.y * 32, c0 = blockIdx.x * 32;
  const int tid = threadIdx.x;
#pragma unroll
  for (int i = 0; i < 4; ++i) {
    int idx = tid + i * 256;
    int rr = idx >> 5, cc = idx & 31;
    sT[cc * 33 + rr] = (__bf16)in[(size_t)(r0 + rr) * Cc + c0 + cc];
  }
  __syncthreads();
#pragma unroll
  for (int i = 0; i < 4; ++i) {
    int idx = tid + i * 256;
    int cc = idx >> 5, rr = idx & 31;
    out[(size_t)(c0 + cc) * R + r0 + rr] = sT[cc * 33 + rr];
  }
}

// ---------------------------------------------------------------------------
// bf16 GEMM, B pre-transposed: out[M,N] = Ab[M,K] * WT[N,K]^T + bias.
// Block 256 thr (8 waves), tile 64(M) x 256(N), K-step 32, double-buffered LDS.
// Wave tile 32x64 -> 8 WMMAs / K-step; all fragments via ds_load_b128.
// SPLIT_V: route columns gn>=2048 (the V part of QKV) to vt[B,H,64,T].
// ---------------------------------------------------------------------------
template <bool BF16_OUT, bool SPLIT_V>
__global__ __launch_bounds__(256) void gemm_bf16wmma(
    const __bf16* __restrict__ Ab, const __bf16* __restrict__ WTb,
    const float* __restrict__ bias, void* __restrict__ outp,
    __bf16* __restrict__ vt, int M, int N, int K) {
  constexpr int BM = 64, BN = 256, BK = 32;
  __shared__ __align__(16) __bf16 sA[2][BM * BK];   // [m][k]
  __shared__ __align__(16) __bf16 sB[2][BN * BK];   // [n][k] N-major

  const int tid = threadIdx.x, wave = tid >> 5, lane = tid & 31;
  const int wm = wave & 1, wn = wave >> 1;
  const int m_blk = blockIdx.y * BM, n_blk = blockIdx.x * BN;
  const int ar = tid >> 2, ac = (tid & 3) * 8;

  auto stage = [&](int k0, int buf) {
    copy16_g2l(&sA[buf][ar * BK + ac], &Ab[(size_t)(m_blk + ar) * K + k0 + ac]);
#pragma unroll
    for (int i = 0; i < 4; ++i) {
      int c = tid + i * 256;
      int n = c >> 2, kk = (c & 3) * 8;
      copy16_g2l(&sB[buf][n * BK + kk], &WTb[(size_t)(n_blk + n) * K + k0 + kk]);
    }
  };

  const f32x8 zero = {0.f, 0.f, 0.f, 0.f, 0.f, 0.f, 0.f, 0.f};
  f32x8 acc[2][4];
#pragma unroll
  for (int i = 0; i < 2; ++i)
#pragma unroll
    for (int j = 0; j < 4; ++j) acc[i][j] = zero;

  stage(0, 0);
  const int nk = K / BK;
  for (int ki = 0; ki < nk; ++ki) {
    wait_async_copies();               // own copies for buf[ki&1] done
    __syncthreads();                   // everyone's copies done + prev compute done
    if (ki + 1 < nk) stage((ki + 1) * BK, (ki + 1) & 1);  // overlap with WMMA below

    const unsigned* sAu = reinterpret_cast<const unsigned*>(sA[ki & 1]);
    const unsigned* sBu = reinterpret_cast<const unsigned*>(sB[ki & 1]);

    Frag a[2], b[4];
#pragma unroll
    for (int i = 0; i < 2; ++i) lds_load_afrag(sAu, wm * 32 + i * 16, BK / 2, 0, a[i]);
#pragma unroll
    for (int j = 0; j < 4; ++j) lds_load_bfrag(sBu, wn * 64 + j * 16, BK / 2, 0, b[j]);
#pragma unroll
    for (int i = 0; i < 2; ++i)
#pragma unroll
      for (int j = 0; j < 4; ++j) acc[i][j] = wmma_bf16(a[i], b[j], acc[i][j]);
  }

  // Epilogue: C element (r, lane) -> M = r + 8*(lane>>4), N = lane&15
#pragma unroll
  for (int i = 0; i < 2; ++i) {
#pragma unroll
    for (int j = 0; j < 4; ++j) {
      int gn = n_blk + wn * 64 + j * 16 + (lane & 15);
      float bv = bias[gn];
#pragma unroll
      for (int r = 0; r < 8; ++r) {
        int gm = m_blk + wm * 32 + i * 16 + r + ((lane >> 4) << 3);
        float val = acc[i][j][r] + bv;
        if (SPLIT_V && gn >= 2048) {
          // V part -> transposed [B,H,64,T]; 8 r's are consecutive t (16B runs)
          int n2 = gn - 2048, hh = n2 >> 6, dd = n2 & 63;
          int bb = gm >> 11, tt = gm & 2047;
          vt[((size_t)(bb * 16 + hh) * 64 + dd) * 2048 + tt] = (__bf16)val;
        } else if (BF16_OUT) {
          ((__bf16*)outp)[(size_t)gm * N + gn] = (__bf16)val;
        } else {
          ((float*)outp)[(size_t)gm * N + gn] = val;
        }
      }
    }
  }
}

// ---------------------------------------------------------------------------
// Causal flash attention: Q,K from bf16 qkv [B,T,3C]; V from vt [B,H,64,T].
// Grid (T/64, H, B), block 128 (4 waves), wave = 16 q-rows.
// Double-buffered K/V tiles; all fragments via ds_load_b128.
// ---------------------------------------------------------------------------
__global__ __launch_bounds__(128) void flash_attn_bf16wmma(
    const __bf16* __restrict__ qkvb, const __bf16* __restrict__ vtb,
    __bf16* __restrict__ Outb) {
  constexpr int T = 2048, Dh = 64, C = 1024, C3 = 3072;
  constexpr float SCL = 0.125f;  // 1/sqrt(64)

  __shared__ __align__(16) __bf16 sQ[64 * 64];      // [q][d]
  __shared__ __align__(16) __bf16 sK[2][64 * 64];   // [kr][d] (N-major for K^T B-frags)
  __shared__ __align__(16) __bf16 sVt[2][64 * 64];  // [d][kr] (N-major for V B-frags)
  __shared__ __align__(16) __bf16 sP[64 * 64];      // [q][kr] C->A layout relay

  const int tid = threadIdx.x, wave = tid >> 5, lane = tid & 31;
  const int qt = blockIdx.x, h = blockIdx.y, b = blockIdx.z;
  const int q0 = qt * 64, m0 = wave * 16;
  const __bf16* vhead = vtb + (size_t)(b * 16 + h) * 64 * 2048;

  // Stage Q (covered by first wait+barrier)
#pragma unroll
  for (int i = 0; i < 4; ++i) {
    int c = tid + i * 128;
    int r = c >> 3, dd = (c & 7) * 8;
    copy16_g2l(&sQ[r * 64 + dd],
               &qkvb[(size_t)(b * T + q0 + r) * C3 + h * Dh + dd]);
  }

  auto stage_kv = [&](int kt, int buf) {
#pragma unroll
    for (int i = 0; i < 4; ++i) {
      int c = tid + i * 128;
      int r = c >> 3, dd = (c & 7) * 8;
      copy16_g2l(&sK[buf][r * 64 + dd],
                 &qkvb[(size_t)(b * T + kt * 64 + r) * C3 + h * Dh + dd + C]);
      // Vt tile: rows are d, columns are kr (contiguous 128B per d row)
      copy16_g2l(&sVt[buf][r * 64 + dd], &vhead[(size_t)r * 2048 + kt * 64 + dd]);
    }
  };

  const f32x8 zero = {0.f, 0.f, 0.f, 0.f, 0.f, 0.f, 0.f, 0.f};
  f32x8 o[4];
#pragma unroll
  for (int dt = 0; dt < 4; ++dt) o[dt] = zero;
  f32x8 mrow, lrow = zero;
#pragma unroll
  for (int r = 0; r < 8; ++r) mrow[r] = -3.0e38f;

  const unsigned* sQu = reinterpret_cast<const unsigned*>(sQ);
  const unsigned* sPu = reinterpret_cast<const unsigned*>(sP);

  stage_kv(0, 0);

  for (int kt = 0; kt <= qt; ++kt) {
    wait_async_copies();                 // own copies for buf[kt&1] done
    __syncthreads();                     // all copies landed + prev compute done
    if (kt < qt) stage_kv(kt + 1, (kt + 1) & 1);  // prefetch into other buffer

    const unsigned* sKu = reinterpret_cast<const unsigned*>(sK[kt & 1]);
    const unsigned* sVu = reinterpret_cast<const unsigned*>(sVt[kt & 1]);

    // S = Q * K^T (unscaled; scale folded into exp below)
    f32x8 s[4];
#pragma unroll
    for (int nt = 0; nt < 4; ++nt) s[nt] = zero;
#pragma unroll
    for (int ks = 0; ks < 2; ++ks) {
      Frag aq; lds_load_afrag(sQu, m0, 32, 16 * ks, aq);
#pragma unroll
      for (int nt = 0; nt < 4; ++nt) {
        Frag bk; lds_load_bfrag(sKu, nt * 16, 32, 16 * ks, bk);
        s[nt] = wmma_bf16(aq, bk, s[nt]);
      }
    }

    // Causal mask on the diagonal tile
    if (kt == qt) {
#pragma unroll
      for (int nt = 0; nt < 4; ++nt)
#pragma unroll
        for (int r = 0; r < 8; ++r) {
          int row = m0 + r + ((lane >> 4) << 3);
          int col = nt * 16 + (lane & 15);
          if (col > row) s[nt][r] = -3.0e38f;
        }
    }

    // Online softmax (max tracked in unscaled domain)
    f32x8 mnew = mrow;
#pragma unroll
    for (int nt = 0; nt < 4; ++nt)
#pragma unroll
      for (int r = 0; r < 8; ++r) mnew[r] = fmaxf(mnew[r], s[nt][r]);
#pragma unroll
    for (int off = 1; off < 16; off <<= 1)
#pragma unroll
      for (int r = 0; r < 8; ++r) mnew[r] = fmaxf(mnew[r], __shfl_xor(mnew[r], off, 16));

    f32x8 alpha, rsum = zero;
#pragma unroll
    for (int r = 0; r < 8; ++r) alpha[r] = __expf(SCL * (mrow[r] - mnew[r]));
    mrow = mnew;

#pragma unroll
    for (int nt = 0; nt < 4; ++nt)
#pragma unroll
      for (int r = 0; r < 8; ++r) {
        float p = __expf(SCL * (s[nt][r] - mnew[r]));
        rsum[r] += p;
        int row = m0 + r + ((lane >> 4) << 3);
        int col = nt * 16 + (lane & 15);
        sP[row * 64 + col] = (__bf16)p;
      }
#pragma unroll
    for (int off = 1; off < 16; off <<= 1)
#pragma unroll
      for (int r = 0; r < 8; ++r) rsum[r] += __shfl_xor(rsum[r], off, 16);
#pragma unroll
    for (int r = 0; r < 8; ++r) lrow[r] = lrow[r] * alpha[r] + rsum[r];
#pragma unroll
    for (int dt = 0; dt < 4; ++dt)
#pragma unroll
      for (int r = 0; r < 8; ++r) o[dt][r] *= alpha[r];

    // O += P * V  (sP is per-wave; DS ops in-order within a wave)
#pragma unroll
    for (int ks = 0; ks < 2; ++ks) {
      Frag ap; lds_load_afrag(sPu, m0, 32, 16 * ks, ap);
#pragma unroll
      for (int dt = 0; dt < 4; ++dt) {
        Frag bv; lds_load_bfrag(sVu, dt * 16, 32, 16 * ks, bv);
        o[dt] = wmma_bf16(ap, bv, o[dt]);
      }
    }
  }

  // Normalize + write bf16 output (feeds out-projection GEMM)
#pragma unroll
  for (int dt = 0; dt < 4; ++dt)
#pragma unroll
    for (int r = 0; r < 8; ++r) {
      int row = q0 + m0 + r + ((lane >> 4) << 3);
      int d   = dt * 16 + (lane & 15);
      Outb[(size_t)(b * T + row) * C + h * Dh + d] = (__bf16)(o[dt][r] / lrow[r]);
    }
}

// ---------------------------------------------------------------------------
extern "C" void kernel_launch(void* const* d_in, const int* in_sizes, int n_in,
                              void* d_out, int out_size, void* d_ws, size_t ws_size,
                              hipStream_t stream) {
  const float* x     = (const float*)d_in[0];
  const float* w_qkv = (const float*)d_in[1];
  const float* b_qkv = (const float*)d_in[2];
  const float* w_out = (const float*)d_in[3];
  const float* b_out = (const float*)d_in[4];
  float* out = (float*)d_out;

  const int Mrows = 2 * 2048;   // B*T = 4096
  const int C = 1024, C3 = 3072;

  // Workspace (bf16): qkv | attn | xb | wqkvT | woutT | vt   (~59 MB)
  __bf16* qkvb  = (__bf16*)d_ws;
  __bf16* attnb = qkvb  + (size_t)Mrows * C3;
  __bf16* xb    = attnb + (size_t)Mrows * C;
  __bf16* wqkvT = xb    + (size_t)Mrows * C;     // [3C][C]
  __bf16* woutT = wqkvT + (size_t)C3 * C;        // [C][C]
  __bf16* vtb   = woutT + (size_t)C * C;         // [B,H,64,T]

  // 0) One-time conversions: x flat; weights transposed-converted
  f32_to_bf16x4<<<dim3((Mrows * C) / 4 / 256), dim3(256), 0, stream>>>(x, xb, (Mrows * C) / 4);
  transpose_f32_to_bf16<<<dim3(C3 / 32, C / 32), dim3(256), 0, stream>>>(w_qkv, wqkvT, C, C3);
  transpose_f32_to_bf16<<<dim3(C / 32, C / 32), dim3(256), 0, stream>>>(w_out, woutT, C, C);

  // 1) QKV projection: Q,K -> qkvb; V -> vtb transposed
  gemm_bf16wmma<true, true><<<dim3(C3 / 256, Mrows / 64), dim3(256), 0, stream>>>(
      xb, wqkvT, b_qkv, qkvb, vtb, Mrows, C3, C);
  // 2) Causal flash attention
  flash_attn_bf16wmma<<<dim3(2048 / 64, 16, 2), dim3(128), 0, stream>>>(qkvb, vtb, attnb);
  // 3) Output projection (fp32 result)
  gemm_bf16wmma<false, false><<<dim3(C / 256, Mrows / 64), dim3(256), 0, stream>>>(
      attnb, woutT, b_out, out, nullptr, Mrows, C, C);
}
